// STSPBlock_54056458387725
// MI455X (gfx1250) — matrix operand order repaired and checked by reading, not compiled, not run on hardware
//
#include <hip/hip_runtime.h>
#include <hip/hip_bf16.h>
#include <math.h>

// ---------------------------------------------------------------------------
// STSP block forward for MI455X (gfx1250, wave32, WMMA + TDM).
// Node convs (3x 64->64 3x3 on 32x32x16) run as implicit GEMM via
// V_WMMA_F32_16X16X32_F16 with weights staged into LDS by the Tensor Data
// Mover (tensor_load_to_lds), A fragments software-pipelined from global.
// ---------------------------------------------------------------------------

typedef _Float16     v16h __attribute__((ext_vector_type(16)));
typedef _Float16     v8h  __attribute__((ext_vector_type(8)));
typedef float        v8f  __attribute__((ext_vector_type(8)));
typedef unsigned int v4u  __attribute__((ext_vector_type(4)));
typedef unsigned int v8u  __attribute__((ext_vector_type(8)));

#define DECAY 0.6f
#define BETA  0.0f

#define WFRAG_HALFS_PER_NODE (9 * 2 * 4 * 32 * 16)  // 36864 halfs = 73728 B

// ---------------- utility kernels ----------------

__global__ void k_zero(float* p, size_t n) {
    size_t i = (size_t)blockIdx.x * blockDim.x + threadIdx.x;
    size_t stride = (size_t)gridDim.x * blockDim.x;
    for (; i < n; i += stride) p[i] = 0.0f;
}

__global__ void k_fill(float* p, float v, int n) {
    int i = blockIdx.x * blockDim.x + threadIdx.x;
    if (i < n) p[i] = v;
}

// Pre-swizzle node-conv weights into per-lane WMMA B-fragment order:
// wfrag[n][tap(9)][ks(2)][ntile(4)][lane(32)][j(16)]  (f16)
// B layout (32x16, KxN): lane half h holds K=8h+j (j<8), K=16+8h+(j-8) (j>=8)
//                        at col=lane&15.
__global__ void k_wfrag(const float* __restrict__ convs_w, _Float16* __restrict__ wfrag) {
    int e = blockIdx.x * blockDim.x + threadIdx.x;
    if (e >= 3 * WFRAG_HALFS_PER_NODE) return;
    int t = e;
    int j    = t & 15;  t >>= 4;
    int lane = t & 31;  t >>= 5;
    int nt   = t & 3;   t >>= 2;
    int ks   = t & 1;   t >>= 1;
    int tap  = t % 9;   t /= 9;
    int n    = t;
    int kk = (j < 8) ? ((lane >> 4) * 8 + j) : (16 + (lane >> 4) * 8 + (j - 8));
    int ci = ks * 32 + kk;
    int co = nt * 16 + (lane & 15);
    float w = convs_w[((size_t)(n * 64 + co) * 64 + ci) * 9 + tap];
    wfrag[e] = (_Float16)w;
}

// conv0 (Cin=2, 3x3, SAME) + BN + stateful LIF + 2x2 avgpool.
// Writes out0 (f32 NCHW) and out0h (f16 NHWC, 1-pixel zero halo [16][34][34][64]).
__global__ void k_conv0(const float* __restrict__ x, int tstep,
                        const float* __restrict__ w0,
                        const float* __restrict__ g, const float* __restrict__ bb,
                        const float* __restrict__ mm, const float* __restrict__ vv,
                        const float* __restrict__ lifw,
                        float* __restrict__ v0, float* __restrict__ out0,
                        _Float16* __restrict__ out0h) {
    int tid = blockIdx.x * blockDim.x + threadIdx.x;
    if (tid >= 16 * 64 * 32 * 32) return;
    int xp = tid & 31, yp = (tid >> 5) & 31, co = (tid >> 10) & 63, b = tid >> 16;
    float sc = g[co] * rsqrtf(vv[co] + 1e-5f);
    float bi = bb[co] - mm[co] * sc;
    float sigw = 1.0f / (1.0f + expf(-lifw[0]));
    const float* x0 = x + (size_t)(tstep * 16 + b) * 2 * 4096;
    float pool = 0.0f;
#pragma unroll
    for (int dy = 0; dy < 2; ++dy) {
#pragma unroll
        for (int dx = 0; dx < 2; ++dx) {
            int yf = 2 * yp + dy, xf = 2 * xp + dx;
            float acc = 0.0f;
#pragma unroll
            for (int ci = 0; ci < 2; ++ci) {
                const float* xc = x0 + ci * 4096;
                const float* wc = w0 + (co * 2 + ci) * 9;
#pragma unroll
                for (int ky = 0; ky < 3; ++ky) {
                    int yy = yf + ky - 1;
                    if (yy < 0 || yy > 63) continue;
#pragma unroll
                    for (int kx = 0; kx < 3; ++kx) {
                        int xx = xf + kx - 1;
                        if (xx < 0 || xx > 63) continue;
                        acc += xc[yy * 64 + xx] * wc[ky * 3 + kx];
                    }
                }
            }
            float z = acc * sc + bi;
            size_t vi = ((size_t)(b * 64 + co) * 64 + yf) * 64 + xf;
            float vm = v0[vi];
            vm = vm + (z - vm) * sigw;
            float sp = (vm >= 1.0f) ? 1.0f : 0.0f;
            v0[vi] = vm * (1.0f - sp);
            pool += sp;
        }
    }
    pool *= 0.25f;
    out0[tid] = pool;
    out0h[((size_t)(b * 34 + yp + 1) * 34 + (xp + 1)) * 64 + co] = (_Float16)pool;
}

// mean over rows of length 1024 (one block per row)
__global__ void k_reduce_mean1024(const float* __restrict__ src, float* __restrict__ dst) {
    __shared__ float red[256];
    size_t r = blockIdx.x;
    const float* p = src + r * 1024;
    float s = 0.0f;
    for (int i = threadIdx.x; i < 1024; i += 256) s += p[i];
    red[threadIdx.x] = s;
    __syncthreads();
    for (int off = 128; off > 0; off >>= 1) {
        if (threadIdx.x < off) red[threadIdx.x] += red[threadIdx.x + off];
        __syncthreads();
    }
    if (threadIdx.x == 0) dst[r] = red[0] * (1.0f / 1024.0f);
}

// Single-block graph math: f0 -> traces row0 -> GAT -> softmax -> S -> prune
// -> diffusion -> op^2 column 0 -> per-(b,node) scales.
__global__ void k_graph(const float* __restrict__ pooled0,
                        const float* __restrict__ ftw, const float* __restrict__ ftb,
                        const float* __restrict__ gatw, const float* __restrict__ gata,
                        float* __restrict__ f0, float* __restrict__ traces,
                        float* __restrict__ S, float* __restrict__ scales) {
    __shared__ float sh_hp[16 * 4 * 64];
    __shared__ float sh_e1[16 * 4 * 4];
    __shared__ float sh_e2[16 * 4 * 4];
    __shared__ float sh_att[16 * 4 * 4];
    __shared__ float sh_Sp[16 * 4 * 4];
    int tid = threadIdx.x;

    for (int e = tid; e < 1024; e += 256) {
        int b = e >> 6, c = e & 63;
        float acc = ftb[c];
        for (int k = 0; k < 64; ++k) acc += pooled0[b * 64 + k] * ftw[c * 64 + k];
        f0[e] = fmaxf(acc, 0.0f);
    }
    __syncthreads();
    for (int e = tid; e < 1024; e += 256) {
        int b = e >> 6, c = e & 63;
        int i = (b * 4 + 0) * 64 + c;
        traces[i] = DECAY * traces[i] + (1.0f - DECAY) * f0[e];
    }
    __syncthreads();
    for (int e = tid; e < 4096; e += 256) {
        int c = e & 63, n = (e >> 6) & 3, b = e >> 8;
        const float* trp = traces + (b * 4 + n) * 64;
        float acc = 0.0f;
        for (int k = 0; k < 64; ++k) acc += trp[k] * gatw[c * 64 + k];
        sh_hp[e] = acc;
    }
    __syncthreads();
    if (tid < 256) {
        int h = tid & 3, n = (tid >> 2) & 3, b = tid >> 4;
        const float* hpp = sh_hp + (b * 4 + n) * 64 + h * 16;
        float a1 = 0.0f, a2 = 0.0f;
        for (int d = 0; d < 16; ++d) {
            a1 += hpp[d] * gata[h * 32 + d];
            a2 += hpp[d] * gata[h * 32 + 16 + d];
        }
        sh_e1[tid] = a1;
        sh_e2[tid] = a2;
    }
    __syncthreads();
    if (tid < 256) {  // (b,i,j) symmetrized mean-head logits / 0.01
        int j = tid & 3, i = (tid >> 2) & 3, b = tid >> 4;
        float s = 0.0f;
        for (int h = 0; h < 4; ++h) {
            float x1 = sh_e1[(b * 4 + i) * 4 + h] + sh_e2[(b * 4 + j) * 4 + h];
            float x2 = sh_e1[(b * 4 + j) * 4 + h] + sh_e2[(b * 4 + i) * 4 + h];
            float l1 = (x1 > 0.0f) ? x1 : 0.2f * x1;
            float l2 = (x2 > 0.0f) ? x2 : 0.2f * x2;
            s += 0.5f * (l1 + l2);
        }
        sh_att[tid] = (s * 0.25f) * 100.0f;
    }
    __syncthreads();
    if (tid < 64) {  // softmax over j; S update
        float* row = sh_att + tid * 4;
        float mx = fmaxf(fmaxf(row[0], row[1]), fmaxf(row[2], row[3]));
        float ex0 = expf(row[0] - mx), ex1 = expf(row[1] - mx);
        float ex2 = expf(row[2] - mx), ex3 = expf(row[3] - mx);
        float inv = 1.0f / (ex0 + ex1 + ex2 + ex3);
        float al[4] = {ex0 * inv, ex1 * inv, ex2 * inv, ex3 * inv};
        for (int j = 0; j < 4; ++j) {
            float sn = BETA * S[tid * 4 + j] + (1.0f - BETA) * al[j];
            S[tid * 4 + j] = sn;
            row[j] = sn;
        }
    }
    __syncthreads();
    if (tid < 64) {  // top-3 prune: keep >= 3rd largest
        float* row = sh_att + tid * 4;
        float arr[4] = {row[0], row[1], row[2], row[3]};
        for (int p = 0; p < 3; ++p)
            for (int q = p + 1; q < 4; ++q)
                if (arr[q] > arr[p]) { float t = arr[p]; arr[p] = arr[q]; arr[q] = t; }
        float kth = arr[2];
        for (int j = 0; j < 4; ++j)
            sh_Sp[tid * 4 + j] = (row[j] >= kth) ? row[j] : 0.0f;
    }
    __syncthreads();
    if (tid < 16) {  // diffusion operator and op^2 column 0
        int b = tid;
        float adj[4][4], dinv[4], op[4][4];
        for (int i = 0; i < 4; ++i)
            for (int j = 0; j < 4; ++j)
                adj[i][j] = 0.5f * (sh_Sp[(b * 4 + i) * 4 + j] + sh_Sp[(b * 4 + j) * 4 + i]);
        for (int i = 0; i < 4; ++i) {
            float s = adj[i][0] + adj[i][1] + adj[i][2] + adj[i][3];
            dinv[i] = rsqrtf(s + 1e-6f);
        }
        for (int i = 0; i < 4; ++i)
            for (int j = 0; j < 4; ++j)
                op[i][j] = dinv[i] * adj[i][j] * dinv[j];
        for (int i = 0; i < 4; ++i) {
            float s = 0.0f;
            for (int k = 0; k < 4; ++k) s += op[i][k] * op[k][0];
            scales[b * 4 + i] = s;  // op^2[b][i][0]
        }
    }
}

// Node convs as implicit GEMM via WMMA, B fragments staged in LDS by the
// Tensor Data Mover. grid = (128 mtiles, 3 nodes), block = 128 (4 waves).
// Wave tile: M=32, N=64 (8 accumulators); A fragments software-pipelined.
__global__ void __launch_bounds__(128)
k_nodeconv(const _Float16* __restrict__ out0h, const _Float16* __restrict__ wfrag,
           const float* __restrict__ scales,
           const float* __restrict__ bg, const float* __restrict__ bbn,
           const float* __restrict__ bm, const float* __restrict__ bv,
           const float* __restrict__ lifsw,
           float* __restrict__ vn, float* __restrict__ sn) {
    extern __shared__ _Float16 lwf[];  // 36864 halfs = 73728 B (node's B frags)
    int n = blockIdx.y;

    // TDM: DMA this node's pre-swizzled weight fragments into LDS.
    // D# per CDNA5 ISA 8.3/8.4: 1-D tile, data_size=8B, 9216 elements.
    if (threadIdx.x < 32) {  // wave 0 issues; TENSORcnt is per-wave
        unsigned int ldsAddr = (unsigned int)(uintptr_t)(&lwf[0]);
        unsigned long long ga =
            (unsigned long long)(uintptr_t)(wfrag + (size_t)n * WFRAG_HALFS_PER_NODE);
        v4u g0;
        g0.x = 1u;                                   // count=1, is_restore=0
        g0.y = ldsAddr;                              // lds_addr (bytes)
        g0.z = (unsigned int)ga;                     // global_addr[31:0]
        g0.w = (unsigned int)((ga >> 32) & 0x01FFFFFFu) | (2u << 30);  // [56:32]|type=2
        v8u g1;
        g1.s0 = 3u << 16;                            // data_size=3 (8B), mask=0
        g1.s1 = (9216u & 0xFFFFu) << 16;             // tensor_dim0[15:0]
        g1.s2 = (9216u >> 16) | (1u << 16);          // tensor_dim0[31:16], tensor_dim1=1
        g1.s3 = (9216u & 0xFFFFu) << 16;             // tile_dim0=9216
        g1.s4 = 1u;                                  // tile_dim1=1, tile_dim2=0
        g1.s5 = 9216u;                               // tensor_dim0_stride[31:0]
        g1.s6 = 0u;
        g1.s7 = 0u;
        asm volatile("tensor_load_to_lds %0, %1" :: "s"(g0), "s"(g1) : "memory");
        __builtin_amdgcn_s_wait_tensorcnt(0);
    }
    __syncthreads();

    int wave = threadIdx.x >> 5, lane = threadIdx.x & 31;
    int mtile = blockIdx.x * 128 + wave * 32;        // wave covers rows [mtile, mtile+32)
    int b = mtile >> 10;
    int lr = lane & 15, hi = lane >> 4;
    int rem0 = (mtile + lr) & 1023;                  // A row, tile 0
    int rem1 = (mtile + 16 + lr) & 1023;             // A row, tile 1
    int y0 = rem0 >> 5, x0 = rem0 & 31;
    int y1 = rem1 >> 5, x1 = rem1 & 31;

    struct APair { v16h a0, a1; };
    auto loadA = [&](int it) -> APair {
        int tap = it >> 1, ks = it & 1;
        int ky = tap / 3, kx = tap - ky * 3;
        size_t koff = (size_t)(hi * 8 + ks * 32);
        const v8h* p0 = (const v8h*)(out0h +
            ((size_t)(b * 34 + y0 + ky) * 34 + (x0 + kx)) * 64 + koff);
        const v8h* p1 = (const v8h*)(out0h +
            ((size_t)(b * 34 + y1 + ky) * 34 + (x1 + kx)) * 64 + koff);
        APair r;
        r.a0 = __builtin_shufflevector(p0[0], p0[2], 0, 1, 2, 3, 4, 5, 6, 7,
                                       8, 9, 10, 11, 12, 13, 14, 15);
        r.a1 = __builtin_shufflevector(p1[0], p1[2], 0, 1, 2, 3, 4, 5, 6, 7,
                                       8, 9, 10, 11, 12, 13, 14, 15);
        return r;
    };

    v8f acc0[4] = {v8f{0}, v8f{0}, v8f{0}, v8f{0}};
    v8f acc1[4] = {v8f{0}, v8f{0}, v8f{0}, v8f{0}};

    APair cur = loadA(0);
    for (int it = 0; it < 18; ++it) {                // it = tap*2 + kstep
        APair nxt = cur;
        if (it < 17) nxt = loadA(it + 1);            // prefetch next A fragments
        const _Float16* wk = lwf + (size_t)it * 2048;
#pragma unroll
        for (int nt = 0; nt < 4; ++nt) {
            v16h bv16 = *(const v16h*)(wk + (size_t)(nt * 32 + lane) * 16);
            acc0[nt] = __builtin_amdgcn_wmma_f32_16x16x32_f16(
                false, cur.a0, false, bv16, (short)0, acc0[nt], false, false);
            acc1[nt] = __builtin_amdgcn_wmma_f32_16x16x32_f16(
                false, cur.a1, false, bv16, (short)0, acc1[nt], false, false);
        }
        cur = nxt;
    }

    // Epilogue: zn = scale * conv * bnscale + bnbias; stateful LIF; spike.
    float scale = scales[b * 4 + n + 1];
    float sigw = 1.0f / (1.0f + expf(-lifsw[n]));
#pragma unroll
    for (int mt = 0; mt < 2; ++mt) {
#pragma unroll
        for (int nt = 0; nt < 4; ++nt) {
            int co = nt * 16 + lr;
            float bsc = bg[n * 64 + co] * rsqrtf(bv[n * 64 + co] + 1e-5f);
            float bbi = bbn[n * 64 + co] - bm[n * 64 + co] * bsc;
            v8f a = mt == 0 ? acc0[nt] : acc1[nt];
            // D layout: lane holds col=lr, rows (mtile + mt*16 + hi*8 + r)
            size_t base = ((size_t)((n * 16 + b) * 64 + co)) * 1024 +
                          (size_t)((mtile & 1023) + mt * 16 + hi * 8);
#pragma unroll
            for (int r = 0; r < 8; ++r) {
                float t = a[r] * scale * bsc + bbi;
                size_t idx = base + r;
                float vm = vn[idx];
                vm = vm + (t - vm) * sigw;
                float sp = (vm >= 1.0f) ? 1.0f : 0.0f;
                vn[idx] = vm * (1.0f - sp);
                sn[idx] = sp;
            }
        }
    }
}

// Single-block: feats_n = relu(pooledN @ ftW^T + ftB); full traces decay-blend.
__global__ void k_feats_traces(const float* __restrict__ pooledN,
                               const float* __restrict__ f0,
                               const float* __restrict__ ftw, const float* __restrict__ ftb,
                               float* __restrict__ traces) {
    __shared__ float featsN[3 * 16 * 64];
    int tid = threadIdx.x;
    for (int e = tid; e < 3072; e += 256) {
        int c = e & 63, nb = e >> 6;
        const float* pp = pooledN + nb * 64;
        float acc = ftb[c];
        for (int k = 0; k < 64; ++k) acc += pp[k] * ftw[c * 64 + k];
        featsN[e] = fmaxf(acc, 0.0f);
    }
    __syncthreads();
    for (int e = tid; e < 4096; e += 256) {
        int c = e & 63, n = (e >> 6) & 3, b = e >> 8;
        float fnew = (n == 0) ? f0[b * 64 + c] : featsN[((n - 1) * 16 + b) * 64 + c];
        int i = (b * 4 + n) * 64 + c;
        traces[i] = DECAY * traces[i] + (1.0f - DECAY) * fnew;
    }
}

// y = sigmoid(w0)*out0 + sum_n sigmoid(w_{n+1})*sn[n]
__global__ void k_output(const float* __restrict__ out0, const float* __restrict__ sn,
                         const float* __restrict__ ow, float* __restrict__ out, int tstep) {
    int tid = blockIdx.x * blockDim.x + threadIdx.x;
    if (tid >= 16 * 64 * 32 * 32) return;
    float r = (1.0f / (1.0f + expf(-ow[0]))) * out0[tid];
#pragma unroll
    for (int n = 0; n < 3; ++n) {
        float wn = 1.0f / (1.0f + expf(-ow[n + 1]));
        r += wn * sn[(size_t)n * 1048576 + tid];
    }
    out[(size_t)tstep * 1048576 + tid] = r;
}

// ---------------------------------------------------------------------------

extern "C" void kernel_launch(void* const* d_in, const int* in_sizes, int n_in,
                              void* d_out, int out_size, void* d_ws, size_t ws_size,
                              hipStream_t stream) {
    const float* x       = (const float*)d_in[0];
    const float* conv0_w = (const float*)d_in[1];
    const float* bn0_g   = (const float*)d_in[2];
    const float* bn0_b   = (const float*)d_in[3];
    const float* bn0_m   = (const float*)d_in[4];
    const float* bn0_v   = (const float*)d_in[5];
    const float* lif0_w  = (const float*)d_in[6];
    const float* convs_w = (const float*)d_in[7];
    const float* bns_g   = (const float*)d_in[8];
    const float* bns_b   = (const float*)d_in[9];
    const float* bns_m   = (const float*)d_in[10];
    const float* bns_v   = (const float*)d_in[11];
    const float* lifs_w  = (const float*)d_in[12];
    const float* ft_w    = (const float*)d_in[13];
    const float* ft_b    = (const float*)d_in[14];
    const float* gat_w   = (const float*)d_in[15];
    const float* gat_a   = (const float*)d_in[16];
    const float* out_wts = (const float*)d_in[17];
    float* out = (float*)d_out;
    float* ws = (float*)d_ws;

    // workspace layout (float units, 256B aligned slots)
    size_t off = 0;
    auto take = [&](size_t nfl) { size_t o = off; off += (nfl + 63) & ~(size_t)63; return o; };
    size_t o_v0   = take(16ull * 64 * 64 * 64);       // LIF0 membrane
    size_t o_vn   = take(3ull * 16 * 64 * 1024);      // node LIF membranes
    size_t o_p0h  = take((16ull * 34 * 34 * 64) / 2); // out0 f16 NHWC + halo
    size_t o_tr   = take(16ull * 4 * 64);             // traces
    size_t o_S    = take(16ull * 16);                 // S
    size_t zeroN  = off;                              // zero-init everything above
    size_t o_out0 = take(16ull * 64 * 1024);
    size_t o_sn   = take(3ull * 16 * 64 * 1024);
    size_t o_pl0  = take(1024);
    size_t o_plN  = take(3072);
    size_t o_f0   = take(1024);
    size_t o_scal = take(64);
    size_t o_wfr  = take((3ull * WFRAG_HALFS_PER_NODE) / 2);
    (void)ws_size; (void)in_sizes; (void)n_in; (void)out_size;

    float*     v0      = ws + o_v0;
    float*     vn      = ws + o_vn;
    _Float16*  out0h   = (_Float16*)(ws + o_p0h);
    float*     traces  = ws + o_tr;
    float*     S       = ws + o_S;
    float*     out0    = ws + o_out0;
    float*     sn      = ws + o_sn;
    float*     pooled0 = ws + o_pl0;
    float*     pooledN = ws + o_plN;
    float*     f0      = ws + o_f0;
    float*     scales  = ws + o_scal;
    _Float16*  wfrag   = (_Float16*)(ws + o_wfr);

    // deterministic state init every call
    k_zero<<<2048, 256, 0, stream>>>(ws, zeroN);
    k_fill<<<1, 256, 0, stream>>>(S, 0.25f, 256);
    k_wfrag<<<(3 * WFRAG_HALFS_PER_NODE) / 256, 256, 0, stream>>>(convs_w, wfrag);

    const size_t ldsBytes = WFRAG_HALFS_PER_NODE * sizeof(_Float16);  // 73728

    for (int t = 0; t < 8; ++t) {
        k_conv0<<<4096, 256, 0, stream>>>(x, t, conv0_w, bn0_g, bn0_b, bn0_m, bn0_v,
                                          lif0_w, v0, out0, out0h);
        k_reduce_mean1024<<<1024, 256, 0, stream>>>(out0, pooled0);
        k_graph<<<1, 256, 0, stream>>>(pooled0, ft_w, ft_b, gat_w, gat_a,
                                       f0, traces, S, scales);
        k_nodeconv<<<dim3(128, 3), 128, ldsBytes, stream>>>(out0h, wfrag, scales,
                                                            bns_g, bns_b, bns_m, bns_v,
                                                            lifs_w, vn, sn);
        k_reduce_mean1024<<<3072, 256, 0, stream>>>(sn, pooledN);
        k_feats_traces<<<1, 256, 0, stream>>>(pooledN, f0, ft_w, ft_b, traces);
        k_output<<<4096, 256, 0, stream>>>(out0, sn, out_wts, out, t);
    }
}
